// Lstm_cell_71880572666161
// MI455X (gfx1250) — compile-verified, matrix operand order checked by reading, not run on hardware
//
#include <hip/hip_runtime.h>
#include <hip/hip_bf16.h>

// ---------------------------------------------------------------------------
// LSTM head for MI455X (gfx1250, wave32, WMMA).
//   T=128, B=256, D=512, H=768, 4H=3072, K = D+H = 1280, OUT=60
// Per-step fused GEMM: gates(256x3072) = [x_t | h] (256x1280 bf16) * W^T,
// W = [W_ih | W_hh] row-major (3072 x 1280) bf16, fp32 accumulation via
// v_wmma_f32_16x16x32_bf16.  K loop split into x-phase / h-phase
// (straight-line addressing) and software-pipelined: fragments for slice
// k+32 are loaded before the 8 WMMAs consuming slice k.
// Wave tile: 32(M) x 64(N) = 8 WMMAs per 32-wide K slice.
// ---------------------------------------------------------------------------

#define TT   128
#define BB   256
#define DD   512
#define HH   768
#define G4H  3072
#define KK   1280   // D + H
#define OUTN 60

typedef __attribute__((ext_vector_type(16))) __bf16 v16bf;
typedef __attribute__((ext_vector_type(8)))  __bf16 v8bf;
typedef __attribute__((ext_vector_type(8)))  float  v8f;

// Load one 16x32 (rows x K) bf16 fragment from a row-major matrix.
// p must already point at (row0 + lane&15)*pitch + kcol + ((lane>>4)<<3).
__device__ __forceinline__ v16bf ldfrag(const __bf16* __restrict__ p) {
    v8bf lo = *(const v8bf*)(p);        // K = kb .. kb+7     (16B)
    v8bf hi = *(const v8bf*)(p + 16);   // K = kb+16 .. kb+23 (16B)
    v16bf f;
#pragma unroll
    for (int i = 0; i < 8; ++i) { f[i] = lo[i]; f[8 + i] = hi[i]; }
    return f;
}

// 2x4 tile of 16x16 WMMAs: acc[i][j] += a_i * b_j
__device__ __forceinline__ void wmma8(const v16bf& a0, const v16bf& a1,
                                      const v16bf& b0, const v16bf& b1,
                                      const v16bf& b2, const v16bf& b3,
                                      v8f (&acc)[2][4]) {
    acc[0][0] = __builtin_amdgcn_wmma_f32_16x16x32_bf16(false, a0, false, b0,
                                                        (short)0, acc[0][0], false, false);
    acc[0][1] = __builtin_amdgcn_wmma_f32_16x16x32_bf16(false, a0, false, b1,
                                                        (short)0, acc[0][1], false, false);
    acc[0][2] = __builtin_amdgcn_wmma_f32_16x16x32_bf16(false, a0, false, b2,
                                                        (short)0, acc[0][2], false, false);
    acc[0][3] = __builtin_amdgcn_wmma_f32_16x16x32_bf16(false, a0, false, b3,
                                                        (short)0, acc[0][3], false, false);
    acc[1][0] = __builtin_amdgcn_wmma_f32_16x16x32_bf16(false, a1, false, b0,
                                                        (short)0, acc[1][0], false, false);
    acc[1][1] = __builtin_amdgcn_wmma_f32_16x16x32_bf16(false, a1, false, b1,
                                                        (short)0, acc[1][1], false, false);
    acc[1][2] = __builtin_amdgcn_wmma_f32_16x16x32_bf16(false, a1, false, b2,
                                                        (short)0, acc[1][2], false, false);
    acc[1][3] = __builtin_amdgcn_wmma_f32_16x16x32_bf16(false, a1, false, b3,
                                                        (short)0, acc[1][3], false, false);
}

// One K-segment of the fused GEMM: acc += A_seg * W_seg^T (software pipelined).
__device__ __forceinline__ void gemm_phase(const __bf16* __restrict__ abase,
                                           int apitch,
                                           const __bf16* __restrict__ wbase,
                                           int mrow, int ncol, int lane,
                                           int kend, v8f (&acc)[2][4]) {
    const int lr  = lane & 15;
    const int kb8 = (lane >> 4) << 3;  // 0 or 8 (ISA 7.12.2 16-bit A layout)

    const __bf16* pa0 = abase + (size_t)(mrow + lr) * apitch + kb8;
    const __bf16* pa1 = pa0 + (size_t)16 * apitch;
    const __bf16* pb  = wbase + (size_t)(ncol + lr) * KK + kb8;

    // prologue: fragments for slice k = 0
    v16bf a0 = ldfrag(pa0);
    v16bf a1 = ldfrag(pa1);
    v16bf b0 = ldfrag(pb);
    v16bf b1 = ldfrag(pb + (size_t)16 * KK);
    v16bf b2 = ldfrag(pb + (size_t)32 * KK);
    v16bf b3 = ldfrag(pb + (size_t)48 * KK);

#pragma unroll 2
    for (int k = 32; k < kend; k += 32) {
        // issue next slice's loads first (overlap with the 8 WMMAs below)
        v16bf na0 = ldfrag(pa0 + k);
        v16bf na1 = ldfrag(pa1 + k);
        v16bf nb0 = ldfrag(pb + k);
        v16bf nb1 = ldfrag(pb + (size_t)16 * KK + k);
        v16bf nb2 = ldfrag(pb + (size_t)32 * KK + k);
        v16bf nb3 = ldfrag(pb + (size_t)48 * KK + k);
        __builtin_prefetch(pb + k + 64, 0, 3);   // warm weight stream

        wmma8(a0, a1, b0, b1, b2, b3, acc);

        a0 = na0; a1 = na1;
        b0 = nb0; b1 = nb1; b2 = nb2; b3 = nb3;
    }
    // epilogue
    wmma8(a0, a1, b0, b1, b2, b3, acc);
}

// ---------------------------------------------------------------------------
// Per-step fused GEMM: gates = [x_t | h_prev] * W^T      (fp32 out)
// grid = (BB/64, G4H/256), block = 256 (8 waves, 2 M-waves x 4 N-waves).
// ---------------------------------------------------------------------------
__global__ __launch_bounds__(256)
void lstm_step_gemm(const __bf16* __restrict__ xt,   // (BB, DD)  bf16, step t
                    const __bf16* __restrict__ hbf,  // (BB, HH)  bf16, h_{t-1}
                    const __bf16* __restrict__ wbf,  // (G4H, KK) bf16
                    float* __restrict__ gates) {     // (BB, G4H) fp32
    const int lane = threadIdx.x & 31;
    const int wave = threadIdx.x >> 5;
    const int wm   = wave & 1;         // 2 M-waves, 32 rows each
    const int wn   = wave >> 1;        // 4 N-waves, 64 cols each

    const int mrow = blockIdx.x * 64 + wm * 32;    // batch rows of C
    const int ncol = blockIdx.y * 256 + wn * 64;   // gate cols of C

    v8f acc[2][4];
#pragma unroll
    for (int i = 0; i < 2; ++i)
#pragma unroll
        for (int j = 0; j < 4; ++j) acc[i][j] = (v8f){};

    // Phase 0: K = 0..511    (x_t, W cols 0..511)
    gemm_phase(xt, DD, wbf, mrow, ncol, lane, DD, acc);
    // Phase 1: K = 512..1279 (h,   W cols 512..1279)
    gemm_phase(hbf, HH, wbf + DD, mrow, ncol, lane, HH, acc);

    // Store C: lane -> N = lane&15, rows (lane>>4)*8 + g  (ISA 7.12.2)
    const int n  = lane & 15;
    const int mb = (lane >> 4) << 3;
#pragma unroll
    for (int i = 0; i < 2; ++i) {
        float* __restrict__ o =
            gates + (size_t)(mrow + i * 16 + mb) * G4H + ncol + n;
#pragma unroll
        for (int g = 0; g < 8; ++g) {
            o[(size_t)g * G4H +  0] = acc[i][0][g];
            o[(size_t)g * G4H + 16] = acc[i][1][g];
            o[(size_t)g * G4H + 32] = acc[i][2][g];
            o[(size_t)g * G4H + 48] = acc[i][3][g];
        }
    }
}

// ---------------------------------------------------------------------------
// Elementwise LSTM cell update (fp32).  idx over BB*HH.
// PyTorch gate order in the 4H axis: [i, f, g, o].
// ---------------------------------------------------------------------------
__device__ __forceinline__ float sigmoidf(float x) {
    return 1.0f / (1.0f + __expf(-x));
}

__global__ __launch_bounds__(256)
void lstm_gate(const float* __restrict__ gates,  // (BB, G4H)
               const float* __restrict__ bsum,   // (G4H)
               float* __restrict__ cst,          // (BB, HH) fp32 state
               __bf16* __restrict__ hbf) {       // (BB, HH) bf16 state
    const int idx = blockIdx.x * blockDim.x + threadIdx.x;
    if (idx >= BB * HH) return;
    const int b = idx / HH;
    const int j = idx - b * HH;
    const float* g0 = gates + (size_t)b * G4H;

    const float gi = sigmoidf(g0[j]          + bsum[j]);
    const float gf = sigmoidf(g0[HH + j]     + bsum[HH + j]);
    const float gg = tanhf   (g0[2 * HH + j] + bsum[2 * HH + j]);
    const float go = sigmoidf(g0[3 * HH + j] + bsum[3 * HH + j]);

    const float cn = gf * cst[idx] + gi * gg;
    cst[idx] = cn;
    hbf[idx] = (__bf16)(go * tanhf(cn));
}

// ---------------------------------------------------------------------------
// Prep kernels (run once per launch; deterministic).
// ---------------------------------------------------------------------------
__global__ __launch_bounds__(256)
void prep_w(const float* __restrict__ wih,  // (G4H, DD)
            const float* __restrict__ whh,  // (G4H, HH)
            __bf16* __restrict__ wbf) {     // (G4H, KK)
    const int idx = blockIdx.x * blockDim.x + threadIdx.x;
    if (idx >= G4H * KK) return;
    const int r = idx / KK;
    const int c = idx - r * KK;
    const float v = (c < DD) ? wih[(size_t)r * DD + c]
                             : whh[(size_t)r * HH + (c - DD)];
    wbf[idx] = (__bf16)v;
}

__global__ __launch_bounds__(256)
void prep_bias(const float* __restrict__ bih, const float* __restrict__ bhh,
               float* __restrict__ bsum) {
    const int i = blockIdx.x * blockDim.x + threadIdx.x;
    if (i < G4H) bsum[i] = bih[i] + bhh[i];
}

__global__ __launch_bounds__(256)
void conv_x(const float* __restrict__ x, __bf16* __restrict__ xbf, int n) {
    const int i = blockIdx.x * blockDim.x + threadIdx.x;
    if (i < n) xbf[i] = (__bf16)x[i];
}

__global__ __launch_bounds__(256)
void init_state(float* __restrict__ cst, __bf16* __restrict__ hbf) {
    const int i = blockIdx.x * blockDim.x + threadIdx.x;
    if (i < BB * HH) { cst[i] = 0.0f; hbf[i] = (__bf16)0.0f; }
}

// ---------------------------------------------------------------------------
// Final FC: out (BB, OUTN) = h_T (bf16) @ W_fc^T + b_fc   (tiny)
// ---------------------------------------------------------------------------
__global__ __launch_bounds__(64)
void fc_head(const __bf16* __restrict__ hbf,   // (BB, HH)
             const float* __restrict__ wfc,    // (OUTN, HH)
             const float* __restrict__ bfc,    // (OUTN)
             float* __restrict__ out) {        // (BB, OUTN)
    const int b = blockIdx.x;
    const int o = threadIdx.x;
    if (o >= OUTN) return;
    const __bf16* h = hbf + (size_t)b * HH;
    const float*  w = wfc + (size_t)o * HH;
    float acc = bfc[o];
#pragma unroll 8
    for (int j = 0; j < HH; ++j) acc = fmaf((float)h[j], w[j], acc);
    out[(size_t)b * OUTN + o] = acc;
}

// ---------------------------------------------------------------------------
// Host launcher
// ---------------------------------------------------------------------------
extern "C" void kernel_launch(void* const* d_in, const int* in_sizes, int n_in,
                              void* d_out, int out_size, void* d_ws, size_t ws_size,
                              hipStream_t stream) {
    const float* X    = (const float*)d_in[0];  // (T, B, D)
    const float* W_ih = (const float*)d_in[1];  // (4H, D)
    const float* W_hh = (const float*)d_in[2];  // (4H, H)
    const float* b_ih = (const float*)d_in[3];  // (4H)
    const float* b_hh = (const float*)d_in[4];  // (4H)
    const float* W_fc = (const float*)d_in[5];  // (OUT, H)
    const float* b_fc = (const float*)d_in[6];  // (OUT)
    float* out = (float*)d_out;                 // (B, OUT)

    // ---- workspace layout (bytes, 256-aligned) ----
    char* ws = (char*)d_ws;
    size_t off = 0;
    auto alloc = [&](size_t bytes) {
        void* p = ws + off;
        off += (bytes + 255) & ~(size_t)255;
        return p;
    };
    __bf16* Xbf   = (__bf16*)alloc((size_t)TT * BB * DD * 2); // 33.5 MB
    __bf16* Wbf   = (__bf16*)alloc((size_t)G4H * KK * 2);     //  7.9 MB
    float*  bsum  = (float*) alloc((size_t)G4H * 4);
    __bf16* Hbf   = (__bf16*)alloc((size_t)BB * HH * 2);
    float*  Cst   = (float*) alloc((size_t)BB * HH * 4);
    float*  gates = (float*) alloc((size_t)BB * G4H * 4);     //  3.1 MB
    (void)ws_size; (void)in_sizes; (void)n_in; (void)out_size;

    // ---- prep: bf16 conversions, bias sum, zero state ----
    {
        const int n = G4H * KK;
        prep_w<<<(n + 255) / 256, 256, 0, stream>>>(W_ih, W_hh, Wbf);
    }
    prep_bias<<<(G4H + 255) / 256, 256, 0, stream>>>(b_ih, b_hh, bsum);
    {
        const int n = TT * BB * DD;
        conv_x<<<(n + 255) / 256, 256, 0, stream>>>(X, Xbf, n);
    }
    init_state<<<(BB * HH + 255) / 256, 256, 0, stream>>>(Cst, Hbf);

    // ---- recurrence: 128 steps, fused [x|h] GEMM + gate update ----
    const dim3 ggrid(BB / 64, G4H / 256);  // (4, 12)
    for (int t = 0; t < TT; ++t) {
        const __bf16* xt = Xbf + (size_t)t * BB * DD;
        lstm_step_gemm<<<ggrid, 256, 0, stream>>>(xt, Hbf, Wbf, gates);
        lstm_gate<<<(BB * HH + 255) / 256, 256, 0, stream>>>(gates, bsum, Cst, Hbf);
    }

    // ---- head ----
    fc_head<<<BB, 64, 0, stream>>>(Hbf, W_fc, b_fc, out);
}
